// MultiHeadConvNNAttention_1683627180139
// MI455X (gfx1250) — matrix operand order repaired and checked by reading, not compiled, not run on hardware
//
#include <hip/hip_runtime.h>
#include <hip/hip_bf16.h>

#define NUM_HEADS 12
#define K_NEIGH   9

typedef __attribute__((ext_vector_type(16))) __bf16 v16bf;
typedef __attribute__((ext_vector_type(8)))  __bf16 v8bf;
typedef __attribute__((ext_vector_type(2)))  __bf16 v2bf;
typedef __attribute__((ext_vector_type(8)))  float  v8f;
typedef __attribute__((ext_vector_type(16))) float  v16f;

static __device__ __forceinline__ v16bf cat8(v8bf lo, v8bf hi) {
  return __builtin_shufflevector(lo, hi, 0,1,2,3,4,5,6,7,8,9,10,11,12,13,14,15);
}

// Branchless descending-sorted top-9 insertion (strict >, keeps earlier index
// on ties when candidates arrive in increasing index order).
static __device__ __forceinline__ void insert9(float (&tv)[9], int (&ti)[9],
                                               float cv, int ci) {
  #pragma unroll
  for (int q = 0; q < 9; ++q) {
    const bool  gt = cv > tv[q];
    const float nv = gt ? cv : tv[q];
    const int   ni = gt ? ci : ti[q];
    cv = gt ? tv[q] : cv;
    ci = gt ? ti[q] : ci;
    tv[q] = nv;  ti[q] = ni;
  }
}

// Tie-aware variant for merging two independently built lists.
static __device__ __forceinline__ void insert9_tie(float (&tv)[9], int (&ti)[9],
                                                   float cv, int ci) {
  #pragma unroll
  for (int q = 0; q < 9; ++q) {
    const bool  gt = (cv > tv[q]) || (cv == tv[q] && ci < ti[q]);
    const float nv = gt ? cv : tv[q];
    const int   ni = gt ? ci : ti[q];
    cv = gt ? tv[q] : cv;
    ci = gt ? ti[q] : ci;
    tv[q] = nv;  ti[q] = ni;
  }
}

// ---------------------------------------------------------------------------
// Kernel 1: L2-normalize each (bh, token) 64-channel vector -> bf16 vn
// ---------------------------------------------------------------------------
__global__ void nn_normalize(const float* __restrict__ x, __bf16* __restrict__ vn) {
  const int wv  = threadIdx.x >> 5;
  const int l   = threadIdx.x & 31;
  const int gid = blockIdx.x * 8 + wv;          // 0 .. 96*1024-1
  const int bh  = gid >> 10;
  const int s   = gid & 1023;
  const int b   = bh / NUM_HEADS;
  const int h   = bh % NUM_HEADS;

  const float* px = x + ((size_t)(b * 1024 + s)) * 768 + h * 64 + l * 2;
  float2 vv = *(const float2*)px;
  float ss = vv.x * vv.x + vv.y * vv.y;
  #pragma unroll
  for (int off = 16; off > 0; off >>= 1) ss += __shfl_xor(ss, off, 32);
  const float rn = 1.0f / fmaxf(sqrtf(ss), 1e-12f);

  v2bf o;
  o[0] = (__bf16)(vv.x * rn);
  o[1] = (__bf16)(vv.y * rn);
  *(v2bf*)(vn + ((size_t)bh << 16) + s * 64 + l * 2) = o;
}

// ---------------------------------------------------------------------------
// Kernel 2: sim = vn^T vn per head via WMMA bf16 (K=64 => 2 wmma / 16x16 tile)
// Entire head's vn (1024 x 64 bf16, rows padded to 144B) staged into LDS with
// CDNA5 async load-to-LDS.  Each wave owns a 16-row strip; streams 64 column
// tiles; per tile stages the 16x16 fp32 scores to LDS (rows padded to 20
// floats so each lane re-reads its 8 candidates with two ds_load_b128),
// computes the block max and usually skips the whole insertion block.
// ---------------------------------------------------------------------------
#define ROW_PAD   144
#define STAGE_OFF (1024 * ROW_PAD)
#define SPAD      20

__global__ void nn_sim_topk(const __bf16* __restrict__ vn, int* __restrict__ topIdx) {
  extern __shared__ char smem[];

  const int bh    = blockIdx.x >> 3;
  const int wgRow = (blockIdx.x & 7) * 128;
  const __bf16* vsrc = vn + ((size_t)bh << 16);

  // Async-stage the full head (1024 rows x 128B) into padded LDS rows.
  for (int i = threadIdx.x; i < 8192; i += 256) {
    const int row = i >> 3, ch = i & 7;
    const uint32_t ldsa = (uint32_t)(uintptr_t)(smem + row * ROW_PAD + ch * 16);
    const uint32_t goff = (uint32_t)(row * 128 + ch * 16);
    asm volatile("global_load_async_to_lds_b128 %0, %1, %2"
                 :: "v"(ldsa), "v"(goff), "s"(vsrc) : "memory");
  }
  asm volatile("s_wait_asynccnt 0x0" ::: "memory");
  __syncthreads();

  const int l  = threadIdx.x & 31;
  const int wv = threadIdx.x >> 5;
  const int g  = l >> 4;       // K-group / column-half of the lane
  const int lr = l & 15;       // row-in-strip / column-in-tile
  const int row = wgRow + wv * 16 + lr;

  // Persistent A fragments for this wave's 16-row strip (K halves 0..31, 32..63)
  const char* rowp = smem + row * ROW_PAD;
  const v16bf A0 = cat8(*(const v8bf*)(rowp + (g * 8) * 2),
                        *(const v8bf*)(rowp + (16 + g * 8) * 2));
  const v16bf A1 = cat8(*(const v8bf*)(rowp + (32 + g * 8) * 2),
                        *(const v8bf*)(rowp + (48 + g * 8) * 2));

  float* stage = (float*)(smem + STAGE_OFF) + wv * 16 * SPAD;

  float tv[9];
  int   ti[9];
  #pragma unroll
  for (int j = 0; j < 9; ++j) { tv[j] = -1.0f; ti[j] = 0; }

  for (int ct = 0; ct < 64; ++ct) {
    const char* colp = smem + (ct * 16 + lr) * ROW_PAD;
    const v16bf B0 = cat8(*(const v8bf*)(colp + (g * 8) * 2),
                          *(const v8bf*)(colp + (16 + g * 8) * 2));
    const v16bf B1 = cat8(*(const v8bf*)(colp + (32 + g * 8) * 2),
                          *(const v8bf*)(colp + (48 + g * 8) * 2));

    v8f c = {};
    c = __builtin_amdgcn_wmma_f32_16x16x32_bf16(false, A0, false, B0, (short)0, c, false, false);
    c = __builtin_amdgcn_wmma_f32_16x16x32_bf16(false, A1, false, B1, (short)0, c, false, false);

    // Stage the 16x16 score tile: lane holds column lr, rows g*8+r.
    #pragma unroll
    for (int r = 0; r < 8; ++r) stage[(g * 8 + r) * SPAD + lr] = c[r];
    __builtin_amdgcn_wave_barrier();
    asm volatile("s_wait_dscnt 0x0" ::: "memory");   // intra-wave LDS RAW

    // Row lr, column half g: 8 candidates via two b128 reads.
    const float4 c0 = *(const float4*)(stage + lr * SPAD + g * 8);
    const float4 c1 = *(const float4*)(stage + lr * SPAD + g * 8 + 4);
    const float bm = fmaxf(fmaxf(fmaxf(c0.x, c0.y), fmaxf(c0.z, c0.w)),
                           fmaxf(fmaxf(c1.x, c1.y), fmaxf(c1.z, c1.w)));
    if (fmaxf(bm, 0.0f) > tv[8]) {        // block-skip: common case rejects all 8
      const float va[8] = {c0.x, c0.y, c0.z, c0.w, c1.x, c1.y, c1.z, c1.w};
      #pragma unroll
      for (int j = 0; j < 8; ++j) {
        const float val = fmaxf(va[j], 0.0f);          // clamp(min=0)
        const int   idx = ct * 16 + g * 8 + j;
        if (val > tv[8]) insert9(tv, ti, val, idx);
      }
    }
    __builtin_amdgcn_wave_barrier();
  }

  // Merge partner half (columns g=1) into lanes 0..15 with index tie-breaking.
  #pragma unroll
  for (int j = 0; j < 9; ++j) {
    const float pv = __shfl_down(tv[j], 16, 32);
    const int   pi = __shfl_down(ti[j], 16, 32);
    if (l < 16) {
      if (pv > tv[8] || (pv == tv[8] && pi < ti[8]))
        insert9_tie(tv, ti, pv, pi);
    }
  }

  if (l < 16) {
    int* op = topIdx + ((size_t)bh * 1024 + row) * 9;
    #pragma unroll
    for (int j = 0; j < 9; ++j) op[j] = ti[j];
  }
}

// ---------------------------------------------------------------------------
// Kernel 3: gather 9 neighbors and contract with conv_w via WMMA.
// K reordered as k = kn*64 + c so every B fragment is 16 CONTIGUOUS floats of
// one gathered neighbor row; conv_w staged once per WG into LDS as bf16.
// ---------------------------------------------------------------------------
#define WPAD 584

__global__ void nn_gather_conv(const float* __restrict__ x,
                               const float* __restrict__ w,
                               const float* __restrict__ bias,
                               const int* __restrict__ topIdx,
                               float* __restrict__ out) {
  extern __shared__ char smem[];
  __bf16* wl = (__bf16*)smem;

  // Re-layout conv_w[o][c][kn] (fp32) -> wl[o][kn*64 + c] (bf16).
  for (int i = threadIdx.x; i < 64 * 576; i += 256) {
    const int o = i / 576, rem = i % 576, c = rem / 9, kn = rem % 9;
    wl[o * WPAD + kn * 64 + c] = (__bf16)w[i];
  }
  __syncthreads();

  const int l  = threadIdx.x & 31;
  const int wv = threadIdx.x >> 5;
  const int g  = l >> 4;
  const int lr = l & 15;
  const int bh  = blockIdx.x >> 3;
  const int grp = (blockIdx.x & 7) * 8 + wv;    // 0..63
  const int b = bh / NUM_HEADS, h = bh % NUM_HEADS;
  const int s = grp * 16 + lr;                  // this lane's token (B column)

  int nbr[9];
  const int* ip = topIdx + ((size_t)bh * 1024 + s) * 9;
  #pragma unroll
  for (int j = 0; j < 9; ++j) nbr[j] = ip[j];

  v8f acc[4] = {};

  for (int kc = 0; kc < 18; ++kc) {
    // B fragment: 16 contiguous channels of one neighbor (k = kn*64 + c).
    const int kbase = kc * 32 + g * 16;
    const int kn = kbase >> 6, c0 = kbase & 63;
    const float* src = x + ((size_t)(b * 1024 + nbr[kn])) * 768 + h * 64 + c0;
    const v16f fv = *(const v16f*)src;
    v16bf Bf;
    #pragma unroll
    for (int e = 0; e < 16; ++e) Bf[e] = (__bf16)fv[e];

    #pragma unroll
    for (int mt = 0; mt < 4; ++mt) {
      const char* rowp = (const char*)(wl + (mt * 16 + lr) * WPAD);
      const v16bf Af = cat8(*(const v8bf*)(rowp + (kc * 32 + g * 16) * 2),
                            *(const v8bf*)(rowp + (kc * 32 + g * 16 + 8) * 2));
      acc[mt] = __builtin_amdgcn_wmma_f32_16x16x32_bf16(false, Af, false, Bf,
                                                        (short)0, acc[mt], false, false);
    }
  }

  // D layout: lane -> column s, 8 consecutive output channels per M-tile.
  float* po = out + ((size_t)(b * 1024 + s)) * 768 + h * 64;
  #pragma unroll
  for (int mt = 0; mt < 4; ++mt) {
    const int o0 = mt * 16 + g * 8;
    float4 r0, r1;
    r0.x = acc[mt][0] + bias[o0 + 0];  r0.y = acc[mt][1] + bias[o0 + 1];
    r0.z = acc[mt][2] + bias[o0 + 2];  r0.w = acc[mt][3] + bias[o0 + 3];
    r1.x = acc[mt][4] + bias[o0 + 4];  r1.y = acc[mt][5] + bias[o0 + 5];
    r1.z = acc[mt][6] + bias[o0 + 6];  r1.w = acc[mt][7] + bias[o0 + 7];
    ((float4*)(po + o0))[0] = r0;
    ((float4*)(po + o0))[1] = r1;
  }
}

// ---------------------------------------------------------------------------
extern "C" void kernel_launch(void* const* d_in, const int* in_sizes, int n_in,
                              void* d_out, int out_size, void* d_ws, size_t ws_size,
                              hipStream_t stream) {
  const float* x      = (const float*)d_in[0];
  const float* conv_w = (const float*)d_in[1];
  const float* conv_b = (const float*)d_in[2];
  float* out = (float*)d_out;

  const size_t vn_bytes  = (size_t)96 * 1024 * 64 * 2;      // 12,582,912
  const size_t idx_bytes = (size_t)96 * 1024 * 9 * 4;       //  3,538,944
  if (ws_size < vn_bytes + idx_bytes) return;

  __bf16* vn  = (__bf16*)d_ws;
  int*    idx = (int*)((char*)d_ws + vn_bytes);

  const size_t lds_sim  = (size_t)1024 * ROW_PAD + 8 * 16 * SPAD * 4;  // 157,696 B
  const size_t lds_conv = (size_t)64 * WPAD * 2;                        //  74,752 B
  (void)hipFuncSetAttribute((const void*)nn_sim_topk,
                            hipFuncAttributeMaxDynamicSharedMemorySize, (int)lds_sim);
  (void)hipFuncSetAttribute((const void*)nn_gather_conv,
                            hipFuncAttributeMaxDynamicSharedMemorySize, (int)lds_conv);

  nn_normalize<<<12288, 256, 0, stream>>>(x, vn);
  nn_sim_topk<<<768, 256, lds_sim, stream>>>(vn, idx);
  nn_gather_conv<<<768, 256, lds_conv, stream>>>(x, conv_w, conv_b, idx, out);
}
